// SequentialTransductionUnitJagged_33500744908964
// MI455X (gfx1250) — matrix operand order repaired and checked by reading, not compiled, not run on hardware
//
#include <hip/hip_runtime.h>
#include <hip/hip_bf16.h>
#include <cstdint>

// ---------------------------------------------------------------------------
// Problem dims (fixed per reference)
#define E   512
#define DL  64
#define DA  64
#define H   8
#define B_  4
#define N_  2048
#define BN  (B_ * N_)          // 8192 rows
#define PROJ_COLS 2048         // DL*2*H + DA*2*H

typedef __attribute__((ext_vector_type(16))) __bf16 v16bf;
typedef __attribute__((ext_vector_type(8)))  __bf16 v8bf;
typedef __attribute__((ext_vector_type(8)))  float  v8f;

// float -> bf16 (round to nearest even) without relying on fp conversion support
static __device__ __forceinline__ __bf16 f2bf(float f) {
    unsigned u = __builtin_bit_cast(unsigned, f);
    unsigned r = (u + 0x7FFFu + ((u >> 16) & 1u)) >> 16;
    unsigned short s = (unsigned short)r;
    return __builtin_bit_cast(__bf16, s);
}

static __device__ __forceinline__ float silu(float x) {
    return x / (1.0f + __expf(-x));
}

// Load a 16x32 bf16 A-fragment (row-major source, K contiguous within a row).
// Per lane: M = lane%16 row; elements 0..7 = K in [half*8, half*8+8),
// elements 8..15 = K in [16+half*8, 16+half*8+8).   (CDNA5 ISA §7.12.2)
static __device__ __forceinline__ v16bf load_a16x32(const __bf16* row, int half) {
    v8bf lo = *(const v8bf*)(row + half * 8);
    v8bf hi = *(const v8bf*)(row + 16 + half * 8);
    return __builtin_shufflevector(lo, hi, 0,1,2,3,4,5,6,7,8,9,10,11,12,13,14,15);
}

static __device__ __forceinline__ v8f wmma_bf16(v16bf a, v16bf b, v8f c) {
    return __builtin_amdgcn_wmma_f32_16x16x32_bf16(false, a, false, b, (short)0, c,
                                                   false, false);
}

// ---------------------------------------------------------------------------
// Kernel 1: layer-norm x -> bf16 normed_x.  One wave per 512-wide row.
__global__ void ln_in_kernel(const float* __restrict__ x, __bf16* __restrict__ nx) {
    int row  = blockIdx.x * 8 + (threadIdx.x >> 5);
    int lane = threadIdx.x & 31;
    const float* xr = x + (size_t)row * E + lane * 16;
    float buf[16];
    float s = 0.f, ss = 0.f;
#pragma unroll
    for (int i = 0; i < 16; ++i) { float v = xr[i]; buf[i] = v; s += v; ss += v * v; }
#pragma unroll
    for (int off = 16; off; off >>= 1) {
        s  += __shfl_xor(s,  off, 32);
        ss += __shfl_xor(ss, off, 32);
    }
    float mean = s * (1.f / E);
    float var  = ss * (1.f / E) - mean * mean;
    float inv  = rsqrtf(var + 1e-6f);
    __bf16* o = nx + (size_t)row * E + lane * 16;
#pragma unroll
    for (int i = 0; i < 16; ++i) o[i] = f2bf((buf[i] - mean) * inv);
}

// Kernel 2: uvqk [E, 2048] f32 -> bf16 transposed [2048, E] (k-contiguous rows)
__global__ void convert_wT_kernel(const float* __restrict__ uvqk, __bf16* __restrict__ wT) {
    int i   = blockIdx.x * blockDim.x + threadIdx.x;   // over 2048*512
    int col = i >> 9;          // / E
    int e   = i & (E - 1);
    wT[i] = f2bf(uvqk[(size_t)e * PROJ_COLS + col]);
}

// Kernel 3: o_w [E, DL*H] f32 -> bf16 (row-major already matches B-frag for @ o_w.T)
__global__ void convert_ow_kernel(const float* __restrict__ ow, __bf16* __restrict__ owb) {
    int i = blockIdx.x * blockDim.x + threadIdx.x;     // over 512*512
    owb[i] = f2bf(ow[i]);
}

// ---------------------------------------------------------------------------
// Kernel 4: projection GEMM  silu(normed_x[8192,512] @ uvqk[512,2048])
// Each wave computes a 16x64 tile (4 WMMA accumulators), K-loop of 16 steps.
// Epilogue routes u->ws(f32), v->out + vT bf16, q->bf16, k->out + bf16.
__global__ void proj_gemm_kernel(const __bf16* __restrict__ A,   // [8192,512]
                                 const __bf16* __restrict__ Bt,  // [2048,512] (n-major)
                                 float*  __restrict__ u,         // [8192,512]
                                 __bf16* __restrict__ qbf,       // [8192,512]
                                 __bf16* __restrict__ kbf,       // [8192,512]
                                 __bf16* __restrict__ vT,        // [B,H,64,N]
                                 float*  __restrict__ kout,      // [8192,512]
                                 float*  __restrict__ vout) {    // [8192,512]
    int wave = threadIdx.x >> 5, lane = threadIdx.x & 31;
    int task = blockIdx.x * 8 + wave;       // 16384 tasks
    int rowTile  = task >> 5;               // 512 row tiles
    int colGroup = task & 31;               // 32 col groups of 64
    int half = lane >> 4, ln16 = lane & 15;

    const __bf16* arow = A + (size_t)(rowTile * 16 + ln16) * E;
    v8f acc[4] = {};
#pragma unroll 4
    for (int k0 = 0; k0 < E; k0 += 32) {
        __builtin_prefetch(arow + k0 + 64, 0, 3);
        v16bf af = load_a16x32(arow + k0, half);
#pragma unroll
        for (int t = 0; t < 4; ++t) {
            int n = colGroup * 64 + t * 16 + ln16;
            v16bf bf_ = load_a16x32(Bt + (size_t)n * E + k0, half);
            acc[t] = wmma_bf16(af, bf_, acc[t]);
        }
    }
#pragma unroll
    for (int t = 0; t < 4; ++t) {
#pragma unroll
        for (int j = 0; j < 8; ++j) {
            int r = rowTile * 16 + j + 8 * half;          // 0..8191
            int c = colGroup * 64 + t * 16 + ln16;        // 0..2047
            float s = silu(acc[t][j]);
            int b = r >> 11, n = r & (N_ - 1);
            if (c < 512) {                                 // u
                u[(size_t)r * 512 + c] = s;
            } else if (c < 1024) {                         // v
                int cc = c - 512;
                vout[(size_t)r * 512 + cc] = s;
                int hh = cc >> 6, dd = cc & 63;
                vT[((size_t)(b * H + hh) * 64 + dd) * N_ + n] = f2bf(s);
            } else if (c < 1536) {                         // q
                qbf[(size_t)r * 512 + (c - 1024)] = f2bf(s);
            } else {                                       // k
                int cc = c - 1536;
                kout[(size_t)r * 512 + cc] = s;
                kbf[(size_t)r * 512 + cc]  = f2bf(s);
            }
        }
    }
}

// ---------------------------------------------------------------------------
// Kernel 5: fused SiLU-attention. Block = (b, 16-row n-tile); 8 waves = 8 heads.
// Per 32-wide m-block: Sᵀ = K·Qᵀ (2x2 WMMA), silu/N * mask (mask tile in LDS,
// shared by all heads), pack accumulators directly into the P A-fragment
// (layout identity, no shuffles), then O += P·V (4 WMMA).
__global__ void attn_kernel(const __bf16* __restrict__ qbf,  // [B,N,H*64]
                            const __bf16* __restrict__ kbf,  // [B,N,H*64]
                            const __bf16* __restrict__ vT,   // [B,H,64,N]
                            const float*  __restrict__ mask, // [B,1,N,N]
                            float* __restrict__ ao) {        // [B,N,H*64]
    __shared__ float smask[16 * 32];
    int h    = threadIdx.x >> 5;
    int lane = threadIdx.x & 31;
    int b    = blockIdx.x >> 7;
    int n0   = (blockIdx.x & 127) * 16;
    int half = lane >> 4, ln16 = lane & 15;

    const __bf16* qrow = qbf + (size_t)(b * N_ + n0 + ln16) * 512 + h * 64;
    v16bf qb0 = load_a16x32(qrow, half);          // kd 0..31
    v16bf qb1 = load_a16x32(qrow + 32, half);     // kd 32..63

    v8f o0 = {}, o1 = {}, o2 = {}, o3 = {};
    const float* mrow = mask + ((size_t)b * N_ + n0) * N_;
    const __bf16* vbase0 = vT + (size_t)(b * H + h) * 64 * N_;

    for (int mb = 0; mb < N_; mb += 32) {
        {   // cooperative mask tile stage: 16x32 f32, reused by all 8 heads
            int idx = threadIdx.x * 2;
            int rr = idx >> 5, cc = idx & 31;
            const float* p = mrow + (size_t)rr * N_ + mb + cc;
            smask[idx]     = p[0];
            smask[idx + 1] = p[1];
        }
        __syncthreads();

        const __bf16* krow0 = kbf + (size_t)(b * N_ + mb + ln16) * 512 + h * 64;
        const __bf16* krow1 = krow0 + (size_t)16 * 512;
        v16bf ka0 = load_a16x32(krow0,      half);
        v16bf ka1 = load_a16x32(krow0 + 32, half);
        v16bf ka2 = load_a16x32(krow1,      half);
        v16bf ka3 = load_a16x32(krow1 + 32, half);

        v8f c0 = {}, c1 = {};
        c0 = wmma_bf16(ka0, qb0, c0);   // Sᵀ rows m..m+15
        c0 = wmma_bf16(ka1, qb1, c0);
        c1 = wmma_bf16(ka2, qb0, c1);   // Sᵀ rows m+16..m+31
        c1 = wmma_bf16(ka3, qb1, c1);

        // silu(S)/N * mask, pack Sᵀ accumulator straight into P A-fragment
        v16bf paf;
#pragma unroll
        for (int j = 0; j < 8; ++j) {
            int ml = j + 8 * half;                       // m offset within tile
            float s0 = silu(c0[j]) * (1.0f / (float)N_) * smask[ln16 * 32 + ml];
            float s1 = silu(c1[j]) * (1.0f / (float)N_) * smask[ln16 * 32 + 16 + ml];
            paf[j]     = f2bf(s0);
            paf[8 + j] = f2bf(s1);
        }

        const __bf16* vb = vbase0 + mb;      // m-contiguous rows
        o0 = wmma_bf16(paf, load_a16x32(vb + (size_t)(ln16)      * N_, half), o0);
        o1 = wmma_bf16(paf, load_a16x32(vb + (size_t)(16 + ln16) * N_, half), o1);
        o2 = wmma_bf16(paf, load_a16x32(vb + (size_t)(32 + ln16) * N_, half), o2);
        o3 = wmma_bf16(paf, load_a16x32(vb + (size_t)(48 + ln16) * N_, half), o3);
        __syncthreads();
    }

    float* orow = ao + (size_t)(b * N_ + n0) * 512 + h * 64;
#pragma unroll
    for (int j = 0; j < 8; ++j) {
        float* p = orow + (size_t)(j + 8 * half) * 512 + ln16;
        p[0]  = o0[j];
        p[16] = o1[j];
        p[32] = o2[j];
        p[48] = o3[j];
    }
}

// ---------------------------------------------------------------------------
// Kernel 6: o_input = u * layer_norm(attn_out) -> bf16
__global__ void ln_mul_kernel(const float* __restrict__ ao, const float* __restrict__ u,
                              __bf16* __restrict__ oin) {
    int row  = blockIdx.x * 8 + (threadIdx.x >> 5);
    int lane = threadIdx.x & 31;
    const float* ar = ao + (size_t)row * E + lane * 16;
    const float* ur = u  + (size_t)row * E + lane * 16;
    float buf[16];
    float s = 0.f, ss = 0.f;
#pragma unroll
    for (int i = 0; i < 16; ++i) { float v = ar[i]; buf[i] = v; s += v; ss += v * v; }
#pragma unroll
    for (int off = 16; off; off >>= 1) {
        s  += __shfl_xor(s,  off, 32);
        ss += __shfl_xor(ss, off, 32);
    }
    float mean = s * (1.f / E);
    float var  = ss * (1.f / E) - mean * mean;
    float inv  = rsqrtf(var + 1e-6f);
    __bf16* o = oin + (size_t)row * E + lane * 16;
#pragma unroll
    for (int i = 0; i < 16; ++i) o[i] = f2bf(ur[i] * (buf[i] - mean) * inv);
}

// ---------------------------------------------------------------------------
// Kernel 7: new_out = o_input @ o_w.T + o_b + x
__global__ void out_gemm_kernel(const __bf16* __restrict__ A,   // [8192,512]
                                const __bf16* __restrict__ Bw,  // o_w bf16 [512,512]
                                const float*  __restrict__ bias,
                                const float*  __restrict__ x,
                                float* __restrict__ out) {
    int wave = threadIdx.x >> 5, lane = threadIdx.x & 31;
    int task = blockIdx.x * 8 + wave;      // 4096 tasks
    int rowTile  = task >> 3;              // 512
    int colGroup = task & 7;               // 8 groups of 64
    int half = lane >> 4, ln16 = lane & 15;

    const __bf16* arow = A + (size_t)(rowTile * 16 + ln16) * E;
    v8f acc[4] = {};
#pragma unroll 4
    for (int k0 = 0; k0 < E; k0 += 32) {
        __builtin_prefetch(arow + k0 + 64, 0, 3);
        v16bf af = load_a16x32(arow + k0, half);
#pragma unroll
        for (int t = 0; t < 4; ++t) {
            int n = colGroup * 64 + t * 16 + ln16;    // output col = row of o_w
            v16bf bf_ = load_a16x32(Bw + (size_t)n * E + k0, half);
            acc[t] = wmma_bf16(af, bf_, acc[t]);
        }
    }
#pragma unroll
    for (int t = 0; t < 4; ++t) {
#pragma unroll
        for (int j = 0; j < 8; ++j) {
            int r = rowTile * 16 + j + 8 * half;
            int c = colGroup * 64 + t * 16 + ln16;
            out[(size_t)r * E + c] = acc[t][j] + bias[c] + x[(size_t)r * E + c];
        }
    }
}

// ---------------------------------------------------------------------------
extern "C" void kernel_launch(void* const* d_in, const int* in_sizes, int n_in,
                              void* d_out, int out_size, void* d_ws, size_t ws_size,
                              hipStream_t stream) {
    const float* x    = (const float*)d_in[0];   // [4,2048,512]
    const float* mask = (const float*)d_in[1];   // [4,1,2048,2048]
    const float* uvqk = (const float*)d_in[2];   // [512,2048]
    const float* o_w  = (const float*)d_in[3];   // [512,512]
    const float* o_b  = (const float*)d_in[4];   // [512]

    float* out_new = (float*)d_out;                       // [4,2048,512]
    float* out_k   = out_new + (size_t)BN * 512;          // [4,2048,512]
    float* out_v   = out_k   + (size_t)BN * 512;          // [4,2048,512]

    char* w = (char*)d_ws;
    __bf16* nx  = (__bf16*)w;  w += (size_t)BN * E * 2;            // 8 MB
    __bf16* wT  = (__bf16*)w;  w += (size_t)PROJ_COLS * E * 2;     // 2 MB
    __bf16* owb = (__bf16*)w;  w += (size_t)E * E * 2;             // 0.5 MB
    __bf16* qbf = (__bf16*)w;  w += (size_t)BN * 512 * 2;          // 8 MB
    __bf16* kbf = (__bf16*)w;  w += (size_t)BN * 512 * 2;          // 8 MB
    __bf16* vT  = (__bf16*)w;  w += (size_t)BN * 512 * 2;          // 8 MB
    __bf16* oin = (__bf16*)w;  w += (size_t)BN * E * 2;            // 8 MB
    float*  u   = (float*)w;   w += (size_t)BN * 512 * 4;          // 16 MB
    float*  ao  = (float*)w;   w += (size_t)BN * 512 * 4;          // 16 MB

    ln_in_kernel<<<BN / 8, 256, 0, stream>>>(x, nx);
    convert_wT_kernel<<<(PROJ_COLS * E) / 256, 256, 0, stream>>>(uvqk, wT);
    convert_ow_kernel<<<(E * E) / 256, 256, 0, stream>>>(o_w, owb);
    proj_gemm_kernel<<<2048, 256, 0, stream>>>(nx, wT, u, qbf, kbf, vT, out_k, out_v);
    attn_kernel<<<B_ * (N_ / 16), 256, 0, stream>>>(qbf, kbf, vT, mask, ao);
    ln_mul_kernel<<<BN / 8, 256, 0, stream>>>(ao, u, oin);
    out_gemm_kernel<<<512, 256, 0, stream>>>(oin, owb, o_b, x, out_new);
}
